// DiscreteMADE_32744830664793
// MI455X (gfx1250) — matrix operand order repaired and checked by reading, not compiled, not run on hardware
//
#include <hip/hip_runtime.h>
#include <hip/hip_bf16.h>

// ---------------------------------------------------------------------------
// DiscreteMADE.log_prob fused kernel for gfx1250 (MI455X, wave32, WMMA).
//
// Sizes: B=65536, K=32 vars, D=32 cats, C=32, H=256, IN=1024, OUT=1024.
// prep kernel: masked weights -> f16, K-major (WMMA B operand rows contiguous).
// made_fused: all 3 layers + log-prob reduction, weights streamed via
// global_load_async_to_lds_b128 (ASYNCcnt) through a triple-buffered LDS tile.
// ---------------------------------------------------------------------------

typedef _Float16 h8   __attribute__((ext_vector_type(8)));
typedef _Float16 v16h __attribute__((ext_vector_type(16)));
typedef float    v8f  __attribute__((ext_vector_type(8)));

#define ASYNC_WAIT_4() asm volatile("s_wait_asynccnt 4" ::: "memory")
#define ASYNC_WAIT_0() asm volatile("s_wait_asynccnt 0" ::: "memory")

// CDNA5 async copy: global -> LDS, 16B per lane, tracked by ASYNCcnt.
__device__ __forceinline__ void async_b128_to_lds(void* lds_dst, const void* gbase,
                                                  unsigned byte_off) {
  // Generic pointer to LDS: low 32 bits are the LDS byte address (ISA 10.2).
  unsigned lds = (unsigned)(unsigned long long)lds_dst;
  unsigned long long gb = (unsigned long long)gbase;
  asm volatile("global_load_async_to_lds_b128 %0, %1, %2"
               :: "v"(lds), "v"(byte_off), "s"(gb) : "memory");
}

// Stage one K-chunk of a K-major f16 weight matrix into LDS:
// 32 k-rows x 256 n, LDS row stride 264 halves (528B, 16B aligned, pads banks).
// 256 threads x 4 async b128 each = 1024 16B segments = 16KB.
__device__ __forceinline__ void stage_chunk(_Float16* dst, const _Float16* src,
                                            int gstride_h, int k0, int n0) {
  int t = threadIdx.x;
#pragma unroll
  for (int s = 0; s < 4; ++s) {
    int seg = t * 4 + s;           // 0..1023
    int row = seg >> 5;            // k-row 0..31
    int col = (seg & 31) << 3;     // n offset in halves, step 8
    async_b128_to_lds(dst + row * 264 + col, src,
                      (unsigned)(((k0 + row) * gstride_h + n0 + col) * 2));
  }
}

// A operand (16x32 f16): lane L -> row L%16, K halves { base..base+7, base+16..base+23 },
// base = (L/16)*8.  rowp points at this lane's row; two 16B LDS loads.
__device__ __forceinline__ v16h lds_load_A(const _Float16* rowp, int baseK) {
  h8 lo = *(const h8*)(rowp + baseK);
  h8 hi = *(const h8*)(rowp + baseK + 16);
  return __builtin_shufflevector(lo, hi, 0, 1, 2, 3, 4, 5, 6, 7,
                                 8, 9, 10, 11, 12, 13, 14, 15);
}

// B operand (32x16 f16, K-major staged): lane L -> column n = t*16 + L%16,
// K rows { base..base+7, base+16..base+23 }, base = (L/16)*8 (same as A).
__device__ __forceinline__ v16h lds_load_B(const _Float16* wb, int lane, int t) {
  v16h b;
  int bk = (lane >> 4) * 8;
#pragma unroll
  for (int s = 0; s < 16; ++s) {
    int k = (s < 8) ? (bk + s) : (bk + 8 + s);
    b[s] = wb[k * 264 + t * 16 + (lane & 15)];
  }
  return b;
}

// 16 N-tiles of D += A x B with a 2-deep register prefetch of B so WMMA issue
// only needs partial s_wait_dscnt credits instead of draining the DS pipe.
__device__ __forceinline__ void gemm_chunk(v8f* acc, const _Float16* wb,
                                           int lane, v16h a) {
  v16h b0 = lds_load_B(wb, lane, 0);
  v16h b1 = lds_load_B(wb, lane, 1);
#pragma unroll
  for (int t = 0; t < 14; ++t) {
    v16h b2 = lds_load_B(wb, lane, t + 2);
    acc[t] = __builtin_amdgcn_wmma_f32_16x16x32_f16(false, a, false, b0,
                                                    (short)0, acc[t], false, false);
    b0 = b1;
    b1 = b2;
  }
  acc[14] = __builtin_amdgcn_wmma_f32_16x16x32_f16(false, a, false, b0,
                                                   (short)0, acc[14], false, false);
  acc[15] = __builtin_amdgcn_wmma_f32_16x16x32_f16(false, a, false, b1,
                                                   (short)0, acc[15], false, false);
}

// ---------------------------------------------------------------------------
// Prep: apply MADE masks, convert to f16, transpose to K-major in workspace.
//   w1t [1024 k][256 n], w2t [256][256], w3t [256 k][1024 n]
// ---------------------------------------------------------------------------
__global__ __launch_bounds__(256) void made_prep(const float* __restrict__ W1,
                                                 const float* __restrict__ W2,
                                                 const float* __restrict__ W3,
                                                 _Float16* __restrict__ ws) {
  int i = blockIdx.x * 256 + threadIdx.x;
  _Float16* w1t = ws;
  _Float16* w2t = ws + 262144;
  _Float16* w3t = ws + 327680;
  if (i < 262144) {                       // W1t: k=i/256 in [0,1024), n=i%256
    int k = i >> 8, n = i & 255;
    int din = (k < 32) ? 0 : (k >> 5);    // deg_in
    float v = ((n & 31) >= din) ? W1[n * 1024 + k] : 0.0f;
    w1t[i] = (_Float16)v;
  } else if (i < 327680) {                // W2t
    int j = i - 262144;
    int k = j >> 8, n = j & 255;
    float v = ((n & 31) >= (k & 31)) ? W2[n * 256 + k] : 0.0f;
    w2t[j] = (_Float16)v;
  } else if (i < 589824) {                // W3t: k=j/1024 in [0,256), n=j%1024
    int j = i - 327680;
    int k = j >> 10, n = j & 1023;
    float v = ((n >> 5) >= (k & 31)) ? W3[n * 256 + k] : 0.0f;
    w3t[j] = (_Float16)v;
  }
}

// ---------------------------------------------------------------------------
// Fused 3-layer MLP + log-prob reduction.  1 WG = 256 thr = 8 waves = 128 rows.
// Triple-buffered async weight staging: per chunk
//   stage(kc+1) -> s_wait_asynccnt 4 -> barrier -> consume(buf kc%3)
// Buffer (kc+1)%3 was last consumed at iteration kc-2, and consume(kc-2)
// precedes barrier(kc-1) which precedes stage(kc+1): safe with ONE barrier.
// ---------------------------------------------------------------------------
__global__ __launch_bounds__(256) void made_fused(
    const int* __restrict__ x, const float* __restrict__ c,
    const float* __restrict__ b1, const float* __restrict__ b2,
    const float* __restrict__ b3,
    const _Float16* __restrict__ w1t, const _Float16* __restrict__ w2t,
    const _Float16* __restrict__ w3t, float* __restrict__ out) {
  __shared__ __align__(16) _Float16 wbuf[3][8448];   // 3 x (32 x 264) weight chunk
  __shared__ __align__(16) _Float16 cbuf[128 * 40];  // c tile, f16, row stride 40
  __shared__ unsigned char xbuf[128 * 32];           // categories
  __shared__ __align__(16) _Float16 hbuf[8][4224];   // per-wave 16 x 264 activ.
  __shared__ float b1s[256], b2s[256], b3s[1024];
  __shared__ float ltile[8][512];                    // per-wave [16 rows][32 vars]

  const int tid = threadIdx.x;
  const int wave = tid >> 5, lane = tid & 31;
  const int lane16 = lane & 15, hihalf = lane >> 4;
  const int baseK = hihalf * 8;
  const int wgrow = blockIdx.x * 128;
  const int rowbase = wave * 16;

  // ---- stage x (u8), c (f16), biases into LDS -----------------------------
  for (int i = tid; i < 128 * 32; i += 256) {
    int r = i >> 5, j = i & 31;
    xbuf[i] = (unsigned char)x[(wgrow + r) * 32 + j];
    cbuf[r * 40 + j] = (_Float16)c[(wgrow + r) * 32 + j];
  }
  b1s[tid] = b1[tid];
  b2s[tid] = b2[tid];
  for (int i = tid; i < 1024; i += 256) b3s[i] = b3[i];

  const _Float16* crow = cbuf + (rowbase + lane16) * 40;
  const unsigned char* xrow = xbuf + (rowbase + lane16) * 32;
  const _Float16* arow = &hbuf[wave][0] + lane16 * 264;

  const v8f vzero = {0.f, 0.f, 0.f, 0.f, 0.f, 0.f, 0.f, 0.f};
  v8f acc[16];

  // =========================== Layer 1 (K = 1024) ==========================
#pragma unroll
  for (int t = 0; t < 16; ++t) acc[t] = vzero;
  stage_chunk(&wbuf[0][0], w1t, 256, 0, 0);
  for (int kc = 0; kc < 32; ++kc) {
    if (kc + 1 < 32) {
      stage_chunk(&wbuf[(kc + 1) % 3][0], w1t, 256, (kc + 1) * 32, 0);
      ASYNC_WAIT_4();
    } else {
      ASYNC_WAIT_0();
    }
    __syncthreads();   // first iteration also publishes x/c/bias staging
    v16h a;
    if (kc == 0) {
      a = lds_load_A(crow, baseK);               // dense condition block
    } else {
      int xv = xrow[kc - 1];                     // one-hot block: build in regs
#pragma unroll
      for (int h = 0; h < 16; ++h) {
        int k = (h < 8) ? (baseK + h) : (baseK + 8 + h);
        a[h] = (xv == k) ? (_Float16)1.0f : (_Float16)0.0f;
      }
    }
    gemm_chunk(acc, &wbuf[kc % 3][0], lane, a);
  }
  // bias + relu + f16, store to per-wave LDS in A-friendly row-major layout
#pragma unroll
  for (int t = 0; t < 16; ++t) {
    float bv = b1s[t * 16 + lane16];
#pragma unroll
    for (int v = 0; v < 8; ++v) {
      float r = acc[t][v] + bv;
      r = r > 0.f ? r : 0.f;
      hbuf[wave][(v + 8 * hihalf) * 264 + t * 16 + lane16] = (_Float16)r;
    }
  }

  // =========================== Layer 2 (K = 256) ===========================
#pragma unroll
  for (int t = 0; t < 16; ++t) acc[t] = vzero;
  stage_chunk(&wbuf[0][0], w2t, 256, 0, 0);
  for (int kc = 0; kc < 8; ++kc) {
    if (kc + 1 < 8) {
      stage_chunk(&wbuf[(kc + 1) % 3][0], w2t, 256, (kc + 1) * 32, 0);
      ASYNC_WAIT_4();
    } else {
      ASYNC_WAIT_0();
    }
    __syncthreads();
    v16h a = lds_load_A(arow + kc * 32, baseK);
    gemm_chunk(acc, &wbuf[kc % 3][0], lane, a);
  }
#pragma unroll
  for (int t = 0; t < 16; ++t) {
    float bv = b2s[t * 16 + lane16];
#pragma unroll
    for (int v = 0; v < 8; ++v) {
      float r = acc[t][v] + bv;
      r = r > 0.f ? r : 0.f;
      hbuf[wave][(v + 8 * hihalf) * 264 + t * 16 + lane16] = (_Float16)r;
    }
  }

  // ================= Layer 3 (K=256, N=1024 in 4 slabs) ====================
  for (int nblk = 0; nblk < 4; ++nblk) {
#pragma unroll
    for (int t = 0; t < 16; ++t) acc[t] = vzero;
    stage_chunk(&wbuf[0][0], w3t, 1024, 0, nblk * 256);
    for (int kc = 0; kc < 8; ++kc) {
      if (kc + 1 < 8) {
        stage_chunk(&wbuf[(kc + 1) % 3][0], w3t, 1024, (kc + 1) * 32, nblk * 256);
        ASYNC_WAIT_4();
      } else {
        ASYNC_WAIT_0();
      }
      __syncthreads();
      v16h a = lds_load_A(arow + kc * 32, baseK);
      gemm_chunk(acc, &wbuf[kc % 3][0], lane, a);
    }
    // consume slab: per (row, var) sum exp over 32 cats (2 WMMA tiles wide)
#pragma unroll
    for (int u = 0; u < 8; ++u) {
      int var = nblk * 8 + u;
      float b3a = b3s[nblk * 256 + u * 32 + lane16];
      float b3b = b3s[nblk * 256 + u * 32 + 16 + lane16];
#pragma unroll
      for (int v = 0; v < 8; ++v) {
        int m = v + 8 * hihalf;
        float y1 = acc[2 * u][v] + b3a;
        float y2 = acc[2 * u + 1][v] + b3b;
        int xm = xbuf[(rowbase + m) * 32 + var];
        float e = __expf(y1) + __expf(y2);
        float ysel = (xm == lane16 ? y1 : 0.f) + (xm == 16 + lane16 ? y2 : 0.f);
#pragma unroll
        for (int ofs = 1; ofs < 16; ofs <<= 1) {
          e += __shfl_xor(e, ofs, 16);
          ysel += __shfl_xor(ysel, ofs, 16);
        }
        if (lane16 == 0) ltile[wave][m * 32 + var] = ysel - __logf(e);
      }
    }
  }

  __syncthreads();
  if (lane < 16) {
    float s = 0.f;
#pragma unroll
    for (int j = 0; j < 32; ++j) s += ltile[wave][lane * 32 + j];
    out[wgrow + rowbase + lane] = s;
  }
}

// ---------------------------------------------------------------------------
extern "C" void kernel_launch(void* const* d_in, const int* in_sizes, int n_in,
                              void* d_out, int out_size, void* d_ws, size_t ws_size,
                              hipStream_t stream) {
  const int*   x  = (const int*)d_in[0];
  const float* c  = (const float*)d_in[1];
  const float* W1 = (const float*)d_in[2];
  const float* b1 = (const float*)d_in[3];
  const float* W2 = (const float*)d_in[4];
  const float* b2 = (const float*)d_in[5];
  const float* W3 = (const float*)d_in[6];
  const float* b3 = (const float*)d_in[7];
  float* out = (float*)d_out;
  _Float16* ws = (_Float16*)d_ws;  // 589824 halves = 1.15 MB (L2-resident)

  made_prep<<<2304, 256, 0, stream>>>(W1, W2, W3, ws);
  made_fused<<<512, 256, 0, stream>>>(x, c, b1, b2, b3,
                                      ws, ws + 262144, ws + 327680, out);
}
